// MMDoubleStreamBlock_82978768158931
// MI455X (gfx1250) — matrix operand order repaired and checked by reading, not compiled, not run on hardware
//
#include <hip/hip_runtime.h>

typedef __bf16 bf16;
typedef __attribute__((ext_vector_type(16))) __bf16 v16bf;
typedef __attribute__((ext_vector_type(8)))  __bf16 v8bf;
typedef __attribute__((ext_vector_type(8)))  float   v8f;

#define LI_   1792
#define LT_   256
#define LTOT_ 2048
#define HID_  3072
#define NH_   24
#define HD_   128
#define MLP_  12288
#define EPS_  1e-6f

// ---------------------------------------------------------------------------
// WMMA fragment helpers (CDNA5 wave32 layouts, CDNA5 ISA §7.12.2)
// ---------------------------------------------------------------------------

// A fragment: 16(M) x 32(K) bf16 tile, row-major source with leading dim ld.
// lane<16 : row=lane,    K = 0..7  and 16..23
// lane>=16: row=lane-16, K = 8..15 and 24..31
__device__ __forceinline__ v16bf load_fragA(const bf16* p, int ld) {
  const int lane = threadIdx.x & 31;
  const int half = lane >> 4;
  const int r    = lane & 15;
  const bf16* row = p + (size_t)r * ld + half * 8;
  v8bf lo = *(const v8bf*)(row);
  v8bf hi = *(const v8bf*)(row + 16);
  v16bf a;
#pragma unroll
  for (int i = 0; i < 8; ++i) { a[i] = lo[i]; a[i + 8] = hi[i]; }
  return a;
}

// B fragment: 32(K) x 16(N) tile loaded from an N-major [N][K] matrix
// (i.e. the transposed weight).  lane<16: col=lane, K=0..7 and 16..23;
// lane>=16: col=lane-16, K=8..15 and 24..31.  Two b128 loads per lane.
__device__ __forceinline__ v16bf load_fragB(const bf16* p, int ld) {
  const int lane = threadIdx.x & 31;
  const int n    = lane & 15;
  const int kh   = (lane >> 4) * 8;
  const bf16* row = p + (size_t)n * ld + kh;
  v8bf lo = *(const v8bf*)(row);
  v8bf hi = *(const v8bf*)(row + 16);
  v16bf b;
#pragma unroll
  for (int i = 0; i < 8; ++i) { b[i] = lo[i]; b[i + 8] = hi[i]; }
  return b;
}

__device__ __forceinline__ v8f wmma_bf16(v16bf a, v16bf b, v8f c) {
  return __builtin_amdgcn_wmma_f32_16x16x32_bf16(false, a, false, b,
                                                 (short)0, c, false, false);
}

__device__ __forceinline__ float gelu_tanh(float x) {
  float x3 = x * x * x;
  return 0.5f * x * (1.0f + tanhf(0.7978845608028654f * (x + 0.044715f * x3)));
}

// ---------------------------------------------------------------------------
// Kernel 0: mod = silu(vec) @ mod_w + mod_b   (tiny GEMV, f32)
// grid.x = 6*HID/256, block = 256
// ---------------------------------------------------------------------------
__global__ void silu_mod_kernel(const float* __restrict__ vec,
                                const float* __restrict__ W,
                                const float* __restrict__ Bb,
                                float* __restrict__ mod) {
  __shared__ float sv[HID_];
  const int tid = threadIdx.x;
  for (int i = tid; i < HID_; i += 256) {
    float v = vec[i];
    sv[i] = v / (1.0f + expf(-v));
  }
  __syncthreads();
  const int j = blockIdx.x * 256 + tid;            // < 6*HID
  float acc = 0.0f;
  for (int k = 0; k < HID_; ++k)
    acc += sv[k] * W[(size_t)k * (6 * HID_) + j];
  mod[j] = acc + Bb[j];
}

// ---------------------------------------------------------------------------
// Kernel 1: transpose + cast f32[K][N] -> bf16[N][K]  (LDS tile transpose)
// block (32,8), grid (N/32, K/32)
// ---------------------------------------------------------------------------
__global__ void transpose_cast_kernel(const float* __restrict__ in,
                                      bf16* __restrict__ out, int K, int N) {
  __shared__ float tile[32][33];
  const int kb = blockIdx.y * 32, nb = blockIdx.x * 32;
  const int tx = threadIdx.x, ty = threadIdx.y;
#pragma unroll
  for (int i = 0; i < 32; i += 8)
    tile[ty + i][tx] = in[(size_t)(kb + ty + i) * N + nb + tx];
  __syncthreads();
#pragma unroll
  for (int i = 0; i < 32; i += 8)
    out[(size_t)(nb + ty + i) * K + kb + tx] = (bf16)tile[tx][ty + i];
}

// ---------------------------------------------------------------------------
// Kernel 2: modulated LayerNorm:  out = LN(x)*(1+scale)+shift  -> bf16 row
// one 256-thread block per token row (HID=3072 -> 12 elems/thread)
// ---------------------------------------------------------------------------
__global__ void ln_mod_kernel(const float* __restrict__ X,
                              bf16* __restrict__ out,
                              const float* __restrict__ shift,
                              const float* __restrict__ scale) {
  const int row = blockIdx.x;
  const int tid = threadIdx.x;
  const float* x = X + (size_t)row * HID_;
  float vals[12];
  float s = 0.0f, ss = 0.0f;
#pragma unroll
  for (int i = 0; i < 12; ++i) {
    float v = x[tid + i * 256];
    vals[i] = v; s += v; ss += v * v;
  }
  __shared__ float r1[256], r2[256];
  r1[tid] = s; r2[tid] = ss;
  __syncthreads();
  for (int st = 128; st > 0; st >>= 1) {
    if (tid < st) { r1[tid] += r1[tid + st]; r2[tid] += r2[tid + st]; }
    __syncthreads();
  }
  const float mean = r1[0] * (1.0f / HID_);
  const float var  = r2[0] * (1.0f / HID_) - mean * mean;
  const float inv  = rsqrtf(var + EPS_);
#pragma unroll
  for (int i = 0; i < 12; ++i) {
    const int c = tid + i * 256;
    float y = (vals[i] - mean) * inv * (1.0f + scale[c]) + shift[c];
    out[(size_t)row * HID_ + c] = (bf16)y;
  }
}

// ---------------------------------------------------------------------------
// Kernel 3: shared WMMA GEMM:  C[M][N] = A_bf16[M][K] @ (BT_bf16[N][K])^T
// block 256 = 8 waves (4M x 2N), wave tile 32x64, block tile 128x128.
// 8 v_wmma per 32-K step against 6 fragment loads (2 A + 4 B).
// mode 0: outF = acc                                   (QKV)
// mode 1: outF = resid + acc*gate[col]                 (attn proj + residual)
// mode 2: outB = gelu(acc + bias[col])                 (fc1)
// mode 3: outF = resid + (acc + bias[col])*gate[col]   (fc2 -> d_out)
// ---------------------------------------------------------------------------
__global__ void gemm_bf16_kernel(const bf16* __restrict__ A,
                                 const bf16* __restrict__ BT,
                                 int M, int N, int K, int mode,
                                 const float* __restrict__ bias,
                                 const float* __restrict__ gate,
                                 const float* __restrict__ resid,
                                 float* __restrict__ outF,
                                 bf16* __restrict__ outB) {
  const int lane = threadIdx.x & 31;
  const int wave = threadIdx.x >> 5;          // 0..7
  const int wm = wave >> 1, wn = wave & 1;    // 4 x 2 wave grid
  const int m0 = blockIdx.y * 128 + wm * 32;
  const int n0 = blockIdx.x * 128 + wn * 64;

  const bf16* Ap = A  + (size_t)m0 * K;
  const bf16* Bp = BT + (size_t)n0 * K;

  v8f accs[2][4] = {};
  for (int k0 = 0; k0 < K; k0 += 32) {
    // Speculative per-lane prefetch ~4 K-steps ahead: lanes cover the 32 rows
    // of the next A tile and both 32-row halves of the B tile.  Addresses past
    // the matrix stay inside the workspace; speculative prefetch drops any
    // translation miss (CDNA5 ISA), no counters touched.
    __builtin_prefetch(Ap + (size_t)lane * K + k0 + 128, 0, 0);
    __builtin_prefetch(Bp + (size_t)lane * K + k0 + 128, 0, 0);
    __builtin_prefetch(Bp + (size_t)(lane + 32) * K + k0 + 128, 0, 0);

    v16bf a[2], b[4];
#pragma unroll
    for (int mi = 0; mi < 2; ++mi)
      a[mi] = load_fragA(Ap + (size_t)(mi * 16) * K + k0, K);
#pragma unroll
    for (int ni = 0; ni < 4; ++ni)
      b[ni] = load_fragB(Bp + (size_t)(ni * 16) * K + k0, K);
#pragma unroll
    for (int mi = 0; mi < 2; ++mi)
#pragma unroll
      for (int ni = 0; ni < 4; ++ni)
        accs[mi][ni] = wmma_bf16(a[mi], b[ni], accs[mi][ni]);
  }

  const int half = lane >> 4, l15 = lane & 15;
#pragma unroll
  for (int mi = 0; mi < 2; ++mi) {
#pragma unroll
    for (int ni = 0; ni < 4; ++ni) {
      const int col = n0 + ni * 16 + l15;
#pragma unroll
      for (int j = 0; j < 8; ++j) {
        const int row = m0 + mi * 16 + half * 8 + j;
        const size_t idx = (size_t)row * N + col;
        const float v = accs[mi][ni][j];
        if (mode == 0) {
          outF[idx] = v;
        } else if (mode == 1) {
          outF[idx] = resid[idx] + v * gate[col];
        } else if (mode == 2) {
          outB[idx] = (bf16)gelu_tanh(v + bias[col]);
        } else {
          outF[idx] = resid[idx] + (v + bias[col]) * gate[col];
        }
      }
    }
  }
}

// ---------------------------------------------------------------------------
// Kernel 4: per-(token, head) RMS-norm of q,k + RoPE (img rows only),
// scatter into attention layouts: qb/kb = [NH][L][HD], vT = [NH][HD][L] bf16
// grid (2048, 24), block 128 (thread = d)
// ---------------------------------------------------------------------------
__global__ void qkv_post_kernel(const float* __restrict__ qkv,
                                const float* __restrict__ qn_img,
                                const float* __restrict__ kn_img,
                                const float* __restrict__ qn_txt,
                                const float* __restrict__ kn_txt,
                                const float* __restrict__ fcos,
                                const float* __restrict__ fsin,
                                bf16* __restrict__ qb, bf16* __restrict__ kb,
                                bf16* __restrict__ vT) {
  const int t = blockIdx.x, h = blockIdx.y, d = threadIdx.x;
  const size_t base = (size_t)t * (3 * HID_) + (size_t)h * HD_;
  const float q = qkv[base + d];
  const float k = qkv[base + HID_ + d];
  const float v = qkv[base + 2 * HID_ + d];

  __shared__ float rq[HD_], rk[HD_];
  rq[d] = q * q; rk[d] = k * k;
  __syncthreads();
  for (int st = 64; st > 0; st >>= 1) {
    if (d < st) { rq[d] += rq[d + st]; rk[d] += rk[d + st]; }
    __syncthreads();
  }
  const bool img = (t < LI_);
  const float* qw = img ? qn_img : qn_txt;
  const float* kw = img ? kn_img : kn_txt;
  float qn = q * rsqrtf(rq[0] * (1.0f / HD_) + EPS_) * qw[d];
  float kn = k * rsqrtf(rk[0] * (1.0f / HD_) + EPS_) * kw[d];
  __syncthreads();

  __shared__ float sq[HD_], sk[HD_];
  sq[d] = qn; sk[d] = kn;
  __syncthreads();
  if (img) {
    const float c = fcos[(size_t)t * HD_ + d];
    const float s = fsin[(size_t)t * HD_ + d];
    const float qr = (d & 1) ? sq[d - 1] : -sq[d + 1];
    const float kr = (d & 1) ? sk[d - 1] : -sk[d + 1];
    qn = qn * c + qr * s;
    kn = kn * c + kr * s;
  }
  const size_t ho = (size_t)h * LTOT_ * HD_;
  qb[ho + (size_t)t * HD_ + d] = (bf16)qn;
  kb[ho + (size_t)t * HD_ + d] = (bf16)kn;
  vT[ho + (size_t)d * LTOT_ + t] = (bf16)v;
}

// ---------------------------------------------------------------------------
// Kernel 5: flash attention, one wave per (head, 16-row q tile).
// S = Q K^T via WMMA (4 k-steps over HD=128), online softmax using the
// C-layout (row = j + 8*half across a 16-lane group), P -> LDS -> A-frag,
// O += P V via WMMA over 32-token key chunks.  Per-head K+V = 1 MB, all-head
// working set 24 MB -> stays resident in the 192 MB L2 across q tiles.
// grid (L/16, NH), block 32
// ---------------------------------------------------------------------------
__global__ void attn_kernel(const bf16* __restrict__ qb,
                            const bf16* __restrict__ kb,
                            const bf16* __restrict__ vT,
                            bf16* __restrict__ attn) {
  const int qt = blockIdx.x;
  const int h  = blockIdx.y;
  const int lane = threadIdx.x;
  const int half = lane >> 4, l15 = lane & 15;

  const bf16* qh = qb + (size_t)h * LTOT_ * HD_ + (size_t)qt * 16 * HD_;
  const bf16* kh = kb + (size_t)h * LTOT_ * HD_;
  const bf16* vh = vT + (size_t)h * HD_ * LTOT_;

  v16bf qf[4];
#pragma unroll
  for (int kk = 0; kk < 4; ++kk) qf[kk] = load_fragA(qh + kk * 32, HD_);

  v8f o[8] = {};
  float mrow[8], lrow[8];
#pragma unroll
  for (int j = 0; j < 8; ++j) { mrow[j] = -1.0e30f; lrow[j] = 0.0f; }

  __shared__ __align__(16) bf16 plds[16][32];
  const float sc = 0.08838834764831845f;   // HD^-0.5

  for (int kt = 0; kt < LTOT_; kt += 32) {
    v8f s0 = {}, s1 = {};
#pragma unroll
    for (int kk = 0; kk < 4; ++kk) {
      s0 = wmma_bf16(qf[kk], load_fragB(kh + (size_t)kt * HD_ + kk * 32, HD_), s0);
      s1 = wmma_bf16(qf[kk], load_fragB(kh + (size_t)(kt + 16) * HD_ + kk * 32, HD_), s1);
    }
#pragma unroll
    for (int j = 0; j < 8; ++j) {
      const float a = s0[j] * sc, b = s1[j] * sc;
      float mx = fmaxf(a, b);
      mx = fmaxf(mx, __shfl_xor(mx, 1));
      mx = fmaxf(mx, __shfl_xor(mx, 2));
      mx = fmaxf(mx, __shfl_xor(mx, 4));
      mx = fmaxf(mx, __shfl_xor(mx, 8));
      const float mnew  = fmaxf(mrow[j], mx);
      const float alpha = __expf(mrow[j] - mnew);
      mrow[j] = mnew;
      lrow[j] *= alpha;
#pragma unroll
      for (int f = 0; f < 8; ++f) o[f][j] *= alpha;
      const float e0 = __expf(a - mnew);
      const float e1 = __expf(b - mnew);
      float sum = e0 + e1;
      sum += __shfl_xor(sum, 1);
      sum += __shfl_xor(sum, 2);
      sum += __shfl_xor(sum, 4);
      sum += __shfl_xor(sum, 8);
      lrow[j] += sum;
      plds[j + 8 * half][l15]      = (bf16)e0;
      plds[j + 8 * half][16 + l15] = (bf16)e1;
    }
    __syncthreads();
    const v16bf pf = load_fragA(&plds[0][0], 32);
#pragma unroll
    for (int f = 0; f < 8; ++f)
      o[f] = wmma_bf16(pf, load_fragB(vh + (size_t)(f * 16) * LTOT_ + kt, LTOT_), o[f]);
    __syncthreads();
  }

  const int colbase = h * HD_;
#pragma unroll
  for (int f = 0; f < 8; ++f)
#pragma unroll
    for (int j = 0; j < 8; ++j) {
      const int row = qt * 16 + half * 8 + j;
      const int col = colbase + f * 16 + l15;
      attn[(size_t)row * HID_ + col] = (bf16)(o[f][j] / lrow[j]);
    }
}

// ---------------------------------------------------------------------------
// Host orchestration
// ---------------------------------------------------------------------------
extern "C" void kernel_launch(void* const* d_in, const int* in_sizes, int n_in,
                              void* d_out, int out_size, void* d_ws, size_t ws_size,
                              hipStream_t stream) {
  (void)in_sizes; (void)n_in; (void)out_size; (void)ws_size;
  const float* img  = (const float*)d_in[0];
  const float* txt  = (const float*)d_in[1];
  const float* vec  = (const float*)d_in[2];
  const float* fcos = (const float*)d_in[3];
  const float* fsin = (const float*)d_in[4];
  const float* mod_w[2]  = {(const float*)d_in[5],  (const float*)d_in[15]};
  const float* mod_b[2]  = {(const float*)d_in[6],  (const float*)d_in[16]};
  const float* qkv_w[2]  = {(const float*)d_in[7],  (const float*)d_in[17]};
  const float* qn_w[2]   = {(const float*)d_in[8],  (const float*)d_in[18]};
  const float* kn_w[2]   = {(const float*)d_in[9],  (const float*)d_in[19]};
  const float* proj_w[2] = {(const float*)d_in[10], (const float*)d_in[20]};
  const float* fc1_w[2]  = {(const float*)d_in[11], (const float*)d_in[21]};
  const float* fc1_b[2]  = {(const float*)d_in[12], (const float*)d_in[22]};
  const float* fc2_w[2]  = {(const float*)d_in[13], (const float*)d_in[23]};
  const float* fc2_b[2]  = {(const float*)d_in[14], (const float*)d_in[24]};

  char* ws = (char*)d_ws;
  size_t off = 0;
  auto take = [&](size_t bytes) -> void* {
    off = (off + 255) & ~(size_t)255;
    void* p = ws + off;
    off += bytes;
    return p;
  };

  float* modv   = (float*)take((size_t)2 * 6 * HID_ * 4);
  bf16* qkvT[2], *projT[2], *fc1T[2], *fc2T[2];
  for (int s = 0; s < 2; ++s) qkvT[s] = (bf16*)take((size_t)3 * HID_ * HID_ * 2);
  for (int s = 0; s < 2; ++s) projT[s] = (bf16*)take((size_t)HID_ * HID_ * 2);
  for (int s = 0; s < 2; ++s) fc1T[s] = (bf16*)take((size_t)MLP_ * HID_ * 2);
  for (int s = 0; s < 2; ++s) fc2T[s] = (bf16*)take((size_t)HID_ * MLP_ * 2);
  bf16*  xm    = (bf16*) take((size_t)LTOT_ * HID_ * 2);   // LN output (both stages)
  float* qkvO  = (float*)take((size_t)LTOT_ * 3 * HID_ * 4);
  bf16*  qb    = (bf16*) take((size_t)NH_ * LTOT_ * HD_ * 2);
  bf16*  kbuf  = (bf16*) take((size_t)NH_ * LTOT_ * HD_ * 2);
  bf16*  vT    = (bf16*) take((size_t)NH_ * HD_ * LTOT_ * 2);
  bf16*  attnb = (bf16*) take((size_t)LTOT_ * HID_ * 2);
  float* hbuf  = (float*)take((size_t)LTOT_ * HID_ * 4);
  bf16*  gbuf  = (bf16*) take((size_t)LTOT_ * MLP_ * 2);

  // 0) modulation vectors
  for (int s = 0; s < 2; ++s)
    silu_mod_kernel<<<dim3(6 * HID_ / 256), 256, 0, stream>>>(
        vec, mod_w[s], mod_b[s], modv + (size_t)s * 6 * HID_);

  // 1) weight casts/transposes to bf16 N-major
  for (int s = 0; s < 2; ++s) {
    transpose_cast_kernel<<<dim3(3 * HID_ / 32, HID_ / 32), dim3(32, 8), 0, stream>>>(
        qkv_w[s], qkvT[s], HID_, 3 * HID_);
    transpose_cast_kernel<<<dim3(HID_ / 32, HID_ / 32), dim3(32, 8), 0, stream>>>(
        proj_w[s], projT[s], HID_, HID_);
    transpose_cast_kernel<<<dim3(MLP_ / 32, HID_ / 32), dim3(32, 8), 0, stream>>>(
        fc1_w[s], fc1T[s], HID_, MLP_);
    transpose_cast_kernel<<<dim3(HID_ / 32, MLP_ / 32), dim3(32, 8), 0, stream>>>(
        fc2_w[s], fc2T[s], MLP_, HID_);
  }

  const float* mI = modv;
  const float* mT = modv + (size_t)6 * HID_;

  // 2) stage-1 modulated LN
  ln_mod_kernel<<<LI_, 256, 0, stream>>>(img, xm, mI + 0 * HID_, mI + 1 * HID_);
  ln_mod_kernel<<<LT_, 256, 0, stream>>>(txt, xm + (size_t)LI_ * HID_,
                                         mT + 0 * HID_, mT + 1 * HID_);

  // 3) QKV GEMMs (mode 0)
  gemm_bf16_kernel<<<dim3(3 * HID_ / 128, LI_ / 128), 256, 0, stream>>>(
      xm, qkvT[0], LI_, 3 * HID_, HID_, 0, nullptr, nullptr, nullptr, qkvO, nullptr);
  gemm_bf16_kernel<<<dim3(3 * HID_ / 128, LT_ / 128), 256, 0, stream>>>(
      xm + (size_t)LI_ * HID_, qkvT[1], LT_, 3 * HID_, HID_, 0,
      nullptr, nullptr, nullptr, qkvO + (size_t)LI_ * 3 * HID_, nullptr);

  // 4) RMS-norm + RoPE + layout for attention
  qkv_post_kernel<<<dim3(LTOT_, NH_), 128, 0, stream>>>(
      qkvO, qn_w[0], kn_w[0], qn_w[1], kn_w[1], fcos, fsin, qb, kbuf, vT);

  // 5) flash attention
  attn_kernel<<<dim3(LTOT_ / 16, NH_), 32, 0, stream>>>(qb, kbuf, vT, attnb);

  // 6) proj + gated residual (mode 1) -> hbuf
  gemm_bf16_kernel<<<dim3(HID_ / 128, LI_ / 128), 256, 0, stream>>>(
      attnb, projT[0], LI_, HID_, HID_, 1, nullptr, mI + 2 * HID_, img, hbuf, nullptr);
  gemm_bf16_kernel<<<dim3(HID_ / 128, LT_ / 128), 256, 0, stream>>>(
      attnb + (size_t)LI_ * HID_, projT[1], LT_, HID_, HID_, 1,
      nullptr, mT + 2 * HID_, txt, hbuf + (size_t)LI_ * HID_, nullptr);

  // 7) stage-2 modulated LN (reuse xm)
  ln_mod_kernel<<<LI_, 256, 0, stream>>>(hbuf, xm, mI + 3 * HID_, mI + 4 * HID_);
  ln_mod_kernel<<<LT_, 256, 0, stream>>>(hbuf + (size_t)LI_ * HID_,
                                         xm + (size_t)LI_ * HID_,
                                         mT + 3 * HID_, mT + 4 * HID_);

  // 8) fc1 + bias + gelu (mode 2) -> gbuf
  gemm_bf16_kernel<<<dim3(MLP_ / 128, LI_ / 128), 256, 0, stream>>>(
      xm, fc1T[0], LI_, MLP_, HID_, 2, fc1_b[0], nullptr, nullptr, nullptr, gbuf);
  gemm_bf16_kernel<<<dim3(MLP_ / 128, LT_ / 128), 256, 0, stream>>>(
      xm + (size_t)LI_ * HID_, fc1T[1], LT_, MLP_, HID_, 2,
      fc1_b[1], nullptr, nullptr, nullptr, gbuf + (size_t)LI_ * MLP_);

  // 9) fc2 + bias + gate + residual (mode 3) -> d_out (img rows then txt rows)
  float* outp = (float*)d_out;
  gemm_bf16_kernel<<<dim3(HID_ / 128, LI_ / 128), 256, 0, stream>>>(
      gbuf, fc2T[0], LI_, HID_, MLP_, 3, fc2_b[0], mI + 5 * HID_, hbuf, outp, nullptr);
  gemm_bf16_kernel<<<dim3(HID_ / 128, LT_ / 128), 256, 0, stream>>>(
      gbuf + (size_t)LI_ * MLP_, fc2T[1], LT_, HID_, MLP_, 3,
      fc2_b[1], mT + 5 * HID_, hbuf + (size_t)LI_ * HID_,
      outp + (size_t)LI_ * HID_, nullptr);
}
